// AttGNN_GCNConv_78503412236426
// MI455X (gfx1250) — matrix operand order repaired and checked by reading, not compiled, not run on hardware
//
#include <hip/hip_runtime.h>

// ---------------------------------------------------------------------------
// AttGNN (4-layer GCN + layer attention + output projection) for gfx1250.
// Dense transforms use v_wmma_f32_16x16x32_bf16 (wave32 WMMA) with both A and
// B fragments fed by contiguous ds_load_b128 (B is staged transposed in LDS,
// weights are pre-transposed at the f32->bf16 convert step).
// Sparse aggregation uses coalesced float4 gathers + global_atomic_add_f32
// into an L2-resident (51MB) aggregation buffer.
// ---------------------------------------------------------------------------

#define HDIM 256
#define CDIM 40

typedef __bf16 bf16_t;
typedef bf16_t v16bf __attribute__((ext_vector_type(16)));
typedef float  v8f   __attribute__((ext_vector_type(8)));

union FragU { uint4 q[2]; v16bf v; };

static __device__ __forceinline__ unsigned short f2bf(float f) {
  union { float f; unsigned u; } v; v.f = f;
  unsigned r = v.u + 0x7FFFu + ((v.u >> 16) & 1u);   // round-to-nearest-even
  return (unsigned short)(r >> 16);
}

// ---------------- degree / normalization -----------------------------------
__global__ void k_deg_init(float* deg, int N) {
  int i = blockIdx.x * blockDim.x + threadIdx.x;
  if (i < N) deg[i] = 1.0f;                     // self loop
}
__global__ void k_deg_count(const long long* __restrict__ dst, float* deg, int E) {
  int e = blockIdx.x * blockDim.x + threadIdx.x;
  if (e < E) atomicAdd(&deg[(int)dst[e]], 1.0f);
}
__global__ void k_dinv(float* deg, int N) {
  int i = blockIdx.x * blockDim.x + threadIdx.x;
  if (i < N) { float d = deg[i]; deg[i] = (d > 0.0f) ? rsqrtf(d) : 0.0f; }
}

// ---------------- f32 -> bf16 convert (plain, for activations) --------------
__global__ void k_f2bf(const float* __restrict__ src, unsigned short* __restrict__ dst,
                       long long count) {
  long long i = (long long)blockIdx.x * blockDim.x + threadIdx.x;
  if (i < count) dst[i] = f2bf(src[i]);
}

// ---------------- f32 -> bf16 convert + transpose (for weights) -------------
// src: W[k][n] (256x256 row-major f32) -> dst: Wt[n][k] (256x256 bf16)
__global__ void k_f2bfT(const float* __restrict__ src, unsigned short* __restrict__ dst) {
  int i = blockIdx.x * blockDim.x + threadIdx.x;    // i = n*256 + k
  if (i < HDIM * HDIM) {
    int n = i >> 8;
    int k = i & 255;
    dst[i] = f2bf(src[k * HDIM + n]);
  }
}

// ---------------- WMMA GEMM: C[N,256] = A[N,256](bf16) @ W[256,256](bf16) ---
// B supplied pre-transposed (Wt[n][k]). Block: 256 threads (8 waves), output
// tile 128 rows x 64 cols; wave w owns row-tile w and iterates 4 col-tiles.
__global__ __launch_bounds__(256) void k_gemm_bf16(
    const unsigned short* __restrict__ A, const unsigned short* __restrict__ Bt,
    float* __restrict__ C, int Nrows) {
  __shared__ unsigned short As[128][32];   // row stride 64B  (16B aligned)
  __shared__ unsigned short Bs[64][40];    // row stride 80B  (16B aligned), [col][k]

  const int tid  = threadIdx.x;
  const int lane = tid & 31;
  const int wave = tid >> 5;
  const int rowBase = blockIdx.x * 128;
  const int colBase = blockIdx.y * 64;

  v8f acc[4];
#pragma unroll
  for (int i = 0; i < 4; ++i)
#pragma unroll
    for (int j = 0; j < 8; ++j) acc[i][j] = 0.0f;

  for (int k0 = 0; k0 < HDIM; k0 += 32) {
    // stage A tile: 128x32 bf16, 16 elems (32B) per thread, coalesced
    {
      int r = tid >> 1;
      int c = (tid & 1) * 16;
      int gr = rowBase + r;
      if (gr < Nrows) {
        const unsigned short* s = A + (size_t)gr * HDIM + k0 + c;
        uint4 p0 = *(const uint4*)(s);
        uint4 p1 = *(const uint4*)(s + 8);
        *(uint4*)&As[r][c]     = p0;
        *(uint4*)&As[r][c + 8] = p1;
      } else {
#pragma unroll
        for (int i = 0; i < 16; ++i) As[r][c + i] = 0;
      }
    }
    // stage B tile (transposed weights): 64 cols x 32 k, 8 elems per thread
    {
      int idx  = tid * 8;
      int rowN = idx >> 5;        // 0..63  (output column)
      int ck   = idx & 31;        // k within tile, multiple of 8
      const unsigned short* s = Bt + (size_t)(colBase + rowN) * HDIM + k0 + ck;
      uint4 p = *(const uint4*)s;
      *(uint4*)&Bs[rowN][ck] = p;
    }
    __syncthreads();

    // A fragment: lanes 0-15 -> K {0..7,16..23}, lanes 16-31 -> K {8..15,24..31}
    FragU af;
    {
      int rl = (wave << 4) + (lane & 15);
      int kb = (lane < 16) ? 0 : 8;
      af.q[0] = *(const uint4*)&As[rl][kb];
      af.q[1] = *(const uint4*)&As[rl][kb + 16];
    }
#pragma unroll
    for (int nt = 0; nt < 4; ++nt) {
      // B fragment: lanes 0-15 hold K {0..7,16..23}, lanes 16-31 hold K {8..15,24..31};
      // column = lane&15. Two ds_load_b128 from Bs.
      FragU bf;
      int cl = (nt << 4) + (lane & 15);
      int kb = (lane < 16) ? 0 : 8;
      bf.q[0] = *(const uint4*)&Bs[cl][kb];
      bf.q[1] = *(const uint4*)&Bs[cl][kb + 16];
      acc[nt] = __builtin_amdgcn_wmma_f32_16x16x32_bf16(
          false, af.v, false, bf.v, (short)0, acc[nt], false, false);
    }
    __syncthreads();
  }

  // store D: row = base + wave*16 + r + 8*(lane>=16), col = base + nt*16 + lane&15
#pragma unroll
  for (int nt = 0; nt < 4; ++nt) {
    int col = colBase + (nt << 4) + (lane & 15);
#pragma unroll
    for (int r = 0; r < 8; ++r) {
      int row = rowBase + (wave << 4) + r + ((lane < 16) ? 0 : 8);
      if (row < Nrows) C[(size_t)row * HDIM + col] = acc[nt][r];
    }
  }
}

// ---------------- aggregation: init with self-loop + bias -------------------
__global__ void k_agg_init(const float* __restrict__ hw, const float* __restrict__ dinv,
                           const float* __restrict__ bias, float* __restrict__ agg, int N) {
  long long idx = (long long)blockIdx.x * blockDim.x + threadIdx.x;  // float4 granules
  long long total = (long long)N * (HDIM / 4);
  if (idx >= total) return;
  int node = (int)(idx >> 6);
  int j4   = (int)(idx & 63);
  float di = dinv[node];
  float s  = di * di;                                  // self-loop norm
  float4 h = ((const float4*)hw)[idx];
  float4 b = ((const float4*)bias)[j4];
  float4 o;
  o.x = h.x * s + b.x; o.y = h.y * s + b.y;
  o.z = h.z * s + b.z; o.w = h.w * s + b.w;
  ((float4*)agg)[idx] = o;
}

// ---------------- edge scatter: agg[dst] += hw[src] * norm ------------------
// 4 edges per 256-thread block; 64 lanes per edge, float4 each (coalesced).
__global__ __launch_bounds__(256) void k_scatter(
    const float* __restrict__ hw, const float* __restrict__ dinv,
    const long long* __restrict__ src, const long long* __restrict__ dst,
    float* __restrict__ agg, int E) {
  int e = blockIdx.x * 4 + (threadIdx.x >> 6);
  if (e >= E) return;
  int t = threadIdx.x & 63;
  int s = (int)src[e];
  int d = (int)dst[e];
  float nrm = dinv[s] * dinv[d];
  float4 v = ((const float4*)(hw + (size_t)s * HDIM))[t];
  float* ap = agg + (size_t)d * HDIM + t * 4;
  atomicAdd(ap + 0, v.x * nrm);
  atomicAdd(ap + 1, v.y * nrm);
  atomicAdd(ap + 2, v.z * nrm);
  atomicAdd(ap + 3, v.w * nrm);
}

// ---------------- relu (in place) + bf16 convert for next layer -------------
__global__ void k_relu_cvt(float* __restrict__ h, unsigned short* __restrict__ hbf,
                           long long count) {
  long long i = (long long)blockIdx.x * blockDim.x + threadIdx.x;
  if (i < count) {
    float v = fmaxf(h[i], 0.0f);
    h[i]   = v;
    hbf[i] = f2bf(v);
  }
}

// ---------------- layer attention + output projection -----------------------
// One wave per node; lane owns channels c = lane + 32*i (coalesced).
__global__ __launch_bounds__(256) void k_att_out(
    const float* __restrict__ h0, const float* __restrict__ h1,
    const float* __restrict__ h2, const float* __restrict__ h3,
    const float* __restrict__ Wout, const float* __restrict__ bout,
    float* __restrict__ out, float* __restrict__ alpha, int N) {
  __shared__ float sh[8][HDIM];
  const int wave = threadIdx.x >> 5;
  const int lane = threadIdx.x & 31;
  const int n = blockIdx.x * 8 + wave;

  if (n < N) {
    const float* ps[4] = { h0 + (size_t)n * HDIM, h1 + (size_t)n * HDIM,
                           h2 + (size_t)n * HDIM, h3 + (size_t)n * HDIM };
    float hl[4][8];
#pragma unroll
    for (int l = 0; l < 4; ++l)
#pragma unroll
      for (int i = 0; i < 8; ++i) hl[l][i] = ps[l][lane + 32 * i];

    float sc[4];
#pragma unroll
    for (int l = 0; l < 4; ++l) {
      float p = 0.0f;
#pragma unroll
      for (int i = 0; i < 8; ++i) p += hl[l][i] * hl[3][i];
      sc[l] = p;
    }
    // butterfly reduce each score across the wave32
#pragma unroll
    for (int off = 16; off > 0; off >>= 1)
#pragma unroll
      for (int l = 0; l < 4; ++l) sc[l] += __shfl_xor(sc[l], off, 32);

    const float scale = 0.0625f;  // 1/sqrt(256)
    float mx = sc[0] * scale;
#pragma unroll
    for (int l = 1; l < 4; ++l) mx = fmaxf(mx, sc[l] * scale);
    float a[4], sum = 0.0f;
#pragma unroll
    for (int l = 0; l < 4; ++l) { a[l] = __expf(sc[l] * scale - mx); sum += a[l]; }
    float inv = 1.0f / sum;
#pragma unroll
    for (int l = 0; l < 4; ++l) a[l] *= inv;

#pragma unroll
    for (int i = 0; i < 8; ++i) {
      float ho = a[0] * hl[0][i] + a[1] * hl[1][i] + a[2] * hl[2][i] + a[3] * hl[3][i];
      sh[wave][lane + 32 * i] = ho;
    }
    if (lane < 4) alpha[(size_t)n * 4 + lane] = a[lane];
  }
  __syncthreads();

  // output projection: 8 nodes x 40 cols staged through LDS
  for (int idx = threadIdx.x; idx < 8 * CDIM; idx += 256) {
    int nl = idx / CDIM;
    int c  = idx % CDIM;
    int gn = blockIdx.x * 8 + nl;
    if (gn < N) {
      float accv = bout[c];
#pragma unroll 8
      for (int k = 0; k < HDIM; ++k) accv += sh[nl][k] * Wout[k * CDIM + c];
      out[(size_t)gn * CDIM + c] = accv;
    }
  }
}

// ---------------------------------------------------------------------------
extern "C" void kernel_launch(void* const* d_in, const int* in_sizes, int n_in,
                              void* d_out, int out_size, void* d_ws, size_t ws_size,
                              hipStream_t stream) {
  const float*     x    = (const float*)d_in[0];
  const long long* ei   = (const long long*)d_in[1];
  const float* W[4] = { (const float*)d_in[2], (const float*)d_in[4],
                        (const float*)d_in[6], (const float*)d_in[8] };
  const float* b[4] = { (const float*)d_in[3], (const float*)d_in[5],
                        (const float*)d_in[7], (const float*)d_in[9] };
  const float* Wout = (const float*)d_in[10];
  const float* bout = (const float*)d_in[11];

  const int N = in_sizes[0] / HDIM;
  const int E = in_sizes[1] / 2;
  const long long NH = (long long)N * HDIM;

  const long long* srcIdx = ei;        // edge_index[0]
  const long long* dstIdx = ei + E;    // edge_index[1]

  // carve workspace (256B aligned)
  char* ws = (char*)d_ws;
  size_t off = 0;
  auto carve = [&](size_t bytes) -> void* {
    void* p = ws + off; off += (bytes + 255) & ~(size_t)255; return p;
  };
  float* dinv = (float*)carve((size_t)N * 4);
  float* hw   = (float*)carve((size_t)NH * 4);
  float* hs[4];
  for (int l = 0; l < 4; ++l) hs[l] = (float*)carve((size_t)NH * 4);
  unsigned short* hbf  = (unsigned short*)carve((size_t)NH * 2);
  unsigned short* wbfT = (unsigned short*)carve((size_t)4 * HDIM * HDIM * 2);

  float* outp   = (float*)d_out;
  float* alphap = outp + (size_t)N * CDIM;

  // degree -> dinv
  k_deg_init<<<(N + 255) / 256, 256, 0, stream>>>(dinv, N);
  k_deg_count<<<(E + 255) / 256, 256, 0, stream>>>(dstIdx, dinv, E);
  k_dinv<<<(N + 255) / 256, 256, 0, stream>>>(dinv, N);

  // bf16 conversions: x (plain) and the 4 layer weights (transposed)
  k_f2bf<<<(int)((NH + 255) / 256), 256, 0, stream>>>(x, hbf, NH);
  for (int l = 0; l < 4; ++l)
    k_f2bfT<<<(HDIM * HDIM + 255) / 256, 256, 0, stream>>>(
        W[l], wbfT + (size_t)l * HDIM * HDIM);

  // 4 GCN layers
  const dim3 gemmGrid((N + 127) / 128, HDIM / 64);
  const int aggBlocks     = (int)(((long long)N * (HDIM / 4) + 255) / 256);
  const int scatterBlocks = (E + 3) / 4;
  const int reluBlocks    = (int)((NH + 255) / 256);
  for (int l = 0; l < 4; ++l) {
    k_gemm_bf16<<<gemmGrid, 256, 0, stream>>>(hbf, wbfT + (size_t)l * HDIM * HDIM, hw, N);
    k_agg_init<<<aggBlocks, 256, 0, stream>>>(hw, dinv, b[l], hs[l], N);
    k_scatter<<<scatterBlocks, 256, 0, stream>>>(hw, dinv, srcIdx, dstIdx, hs[l], E);
    k_relu_cvt<<<reluBlocks, 256, 0, stream>>>(hs[l], hbf, NH);
  }

  // attention over layers + output projection
  k_att_out<<<(N + 7) / 8, 256, 0, stream>>>(hs[0], hs[1], hs[2], hs[3],
                                             Wout, bout, outp, alphap, N);
}
